// GMHAtt_57836029608150
// MI455X (gfx1250) — matrix operand order repaired and checked by reading, not compiled, not run on hardware
//
#include <hip/hip_runtime.h>

// ---------------------------------------------------------------------------
// Gated multi-head attention for MI455X (gfx1250, wave32, WMMA + TDM).
// Heavy contractions: v_wmma_f32_16x16x32_f16 (f32 accumulate).
// f16 tile staging: tensor_load_to_lds (TDM) with LDS row padding.
// ---------------------------------------------------------------------------

typedef __attribute__((ext_vector_type(16))) _Float16 v16h;
typedef __attribute__((ext_vector_type(8)))  _Float16 v8h;
typedef __attribute__((ext_vector_type(8)))  float    v8f;
typedef unsigned int u32x4 __attribute__((ext_vector_type(4)));
typedef int          i32x4 __attribute__((ext_vector_type(4)));
typedef int          i32x8 __attribute__((ext_vector_type(8)));

union F16x16 { v16h v; v8h h8[2]; _Float16 h[16]; };
union F8f    { v8f  v; float f[8]; };

__device__ __forceinline__ v8f wmma_f16(v16h a, v16h b, v8f c) {
    // 8 args: (neg_a, A, neg_b, B, c_mod, C, reuse_a, reuse_b)
    return __builtin_amdgcn_wmma_f32_16x16x32_f16(false, a, false, b,
                                                  (short)0, c, false, false);
}

// ---------------------------------------------------------------------------
// TDM: DMA a 2D f16 tile (rows x width elements, row stride `stride` elements)
// from global memory into LDS at byte offset lds_off, inserting LDS padding
// every 2^(pad_int_enc+1) DWORDs (i.e. per row) so the LDS pitch matches our
// padded shared tiles. Issued by one wave; completion via TENSORcnt.
// D# packing per CDNA5 ISA ch.8 (group0/group1, 2D tile).
// This toolchain's builtin arity: (u32x4, i32x8, i32x4, i32x4, i32x8, i32).
// ---------------------------------------------------------------------------
__device__ __forceinline__ void tdm_load_tile_f16(unsigned lds_off,
                                                  unsigned long long gaddr,
                                                  unsigned width, unsigned rows,
                                                  unsigned stride,
                                                  unsigned pad_int_enc,
                                                  unsigned pad_amt_enc) {
    u32x4 g0;
    g0[0] = 1u;                                        // count=1 (valid user D#)
    g0[1] = lds_off;                                   // lds_addr
    g0[2] = (unsigned)gaddr;                           // global_addr[31:0]
    g0[3] = (unsigned)((gaddr >> 32) & 0x01ffffffu)    // global_addr[56:32]
            | (2u << 30);                              // type=2 ("image")
    i32x8 g1;
    g1[0] = (int)((1u << 16)                           // data_size: 2 bytes
                  | (1u << 20)                         // pad_enable
                  | (pad_int_enc << 22)                // pad_interval
                  | (pad_amt_enc << 25));              // pad_amount
    g1[1] = (int)(width << 16);                        // tensor_dim0[15:0]
    g1[2] = (int)(rows << 16);                         // tensor_dim1[15:0]
    g1[3] = (int)(width << 16);                        // tile_dim0
    g1[4] = (int)rows;                                 // tile_dim1 (tile_dim2=0)
    g1[5] = (int)stride;                               // tensor_dim0_stride[31:0]
    g1[6] = 0;
    g1[7] = 0;
    const i32x4 z4 = {0, 0, 0, 0};
    const i32x8 z8 = {0, 0, 0, 0, 0, 0, 0, 0};
    __builtin_amdgcn_tensor_load_to_lds(g0, g1, z4, z4, z8, 0);
}

// Problem constants (match the reference).
constexpr int Bc  = 32;
constexpr int Sc  = 512;
constexpr int Hc  = 1024;
constexpr int NHc = 8;
constexpr int DHc = Hc / NHc;       // 128
constexpr int MS  = Bc * Sc;        // 16384 tokens

// ---------------------------------------------------------------------------
// Tiled WMMA GEMM: C[M,N] = A[M,K] @ W[K,N] + bias[N]
//   A: fp32 or f16 (A_HALF), W/bias: fp32, C: f16 or fp32 (OUT_HALF)
// Block: 256 threads = 8 waves; 128x128 block tile; wave grid 4x2 each 32x64.
// Double-buffered LDS (ping-pong), K step 32. f16 A tiles staged via TDM.
// ---------------------------------------------------------------------------
template <bool A_HALF, bool OUT_HALF>
__launch_bounds__(256)
__global__ void gemm_bias_wmma(const void* __restrict__ Av,
                               const float* __restrict__ W,
                               const float* __restrict__ bias,
                               void* __restrict__ Cv,
                               int M, int K, int N) {
    __shared__ _Float16 As[2][128][40];   // [buf][row][k], 16B-aligned rows
    __shared__ _Float16 Bs[2][128][40];   // [buf][n][k]   (W tile transposed)

    const int tid  = threadIdx.x;
    const int lane = tid & 31;
    const int wave = tid >> 5;
    const int g    = lane >> 4;        // half-wave (0/1)
    const int lr   = lane & 15;
    const int waveRow = wave & 3;      // 0..3 -> 32-row strip
    const int waveCol = wave >> 2;     // 0..1 -> 64-col strip
    const int rowBase = blockIdx.x * 128;
    const int colBase = blockIdx.y * 128;

    const float*    Af = (const float*)Av;
    const _Float16* Ah = (const _Float16*)Av;

    auto stage = [&](int buf, int k0) {
        // ---- A tile (128x32) ----
        if constexpr (A_HALF) {
            // f16 source: one TDM descriptor moves the whole tile, padding
            // each 32-half row out to the 40-half LDS pitch.
            if (wave == 0) {
                const unsigned lds_off = (unsigned)(size_t)&As[buf][0][0];
                const unsigned long long ga =
                    (unsigned long long)(size_t)(Ah + (size_t)rowBase * K + k0);
                // row = 32 halves = 16 DW -> pad_interval enc 3 (2^4 DW)
                // pad  = 8 halves = 4 DW  -> pad_amount enc 3
                tdm_load_tile_f16(lds_off, ga, 32u, 128u, (unsigned)K, 3u, 3u);
            }
        } else {
            for (int i = tid; i < 128 * 32; i += 256) {
                const int r = i >> 5, c = i & 31;
                As[buf][r][c] =
                    (_Float16)Af[(size_t)(rowBase + r) * K + k0 + c];
            }
            if (tid < 128)
                __builtin_prefetch(&Af[(size_t)(rowBase + tid) * K + k0 + 32], 0, 1);
        }
        // ---- W tile transposed (32x128 -> [n][k]), fp32->f16 on VALU ----
        for (int i = tid; i < 32 * 128; i += 256) {
            const int kk = i >> 7, n = i & 127;
            Bs[buf][n][kk] = (_Float16)W[(size_t)(k0 + kk) * N + colBase + n];
        }
        if (tid < 32)
            __builtin_prefetch(&W[(size_t)(k0 + 32 + tid) * N + colBase], 0, 1);
    };

    const v8f vzero = {0.f, 0.f, 0.f, 0.f, 0.f, 0.f, 0.f, 0.f};
    v8f acc[2][4];
    for (int rt = 0; rt < 2; ++rt)
        for (int ct = 0; ct < 4; ++ct) acc[rt][ct] = vzero;

    const int nk = K / 32;
    stage(0, 0);
    if constexpr (A_HALF) {
        if (wave == 0) __builtin_amdgcn_s_wait_tensorcnt(0);
    }
    __syncthreads();

    for (int kt = 0; kt < nk; ++kt) {
        const int cur = kt & 1;
        if (kt + 1 < nk) stage(cur ^ 1, (kt + 1) * 32);

        // ---- A fragments: lane holds row (lane%16); halves 0..7 = k(8g+0..7),
        //      halves 8..15 = k(8g+16..23)  (ISA 16-bit A 16x32 layout) ----
        F16x16 afrag[2];
        for (int rt = 0; rt < 2; ++rt) {
            const _Float16* src = &As[cur][waveRow * 32 + rt * 16 + lr][0];
            afrag[rt].h8[0] = *(const v8h*)(src + 8 * g);
            afrag[rt].h8[1] = *(const v8h*)(src + 8 * g + 16);
        }
        // ---- B fragments + WMMA ----
        for (int ct = 0; ct < 4; ++ct) {
            F16x16 bfrag;
            const _Float16* src = &Bs[cur][waveCol * 64 + ct * 16 + lr][0];
            bfrag.h8[0] = *(const v8h*)(src + 8 * g);
            bfrag.h8[1] = *(const v8h*)(src + 8 * g + 16);
            for (int rt = 0; rt < 2; ++rt)
                acc[rt][ct] = wmma_f16(afrag[rt].v, bfrag.v, acc[rt][ct]);
        }
        if constexpr (A_HALF) {
            if (wave == 0) __builtin_amdgcn_s_wait_tensorcnt(0);
        }
        __syncthreads();
    }

    // ---- epilogue: bias add, store ----
    for (int rt = 0; rt < 2; ++rt) {
        for (int ct = 0; ct < 4; ++ct) {
            const int col = colBase + waveCol * 64 + ct * 16 + lr;
            const float bval = bias[col];
            F8f a; a.v = acc[rt][ct];
            for (int r = 0; r < 8; ++r) {
                const int row = rowBase + waveRow * 32 + rt * 16 + r + 8 * g;
                const float val = a.f[r] + bval;
                if constexpr (OUT_HALF)
                    ((_Float16*)Cv)[(size_t)row * N + col] = (_Float16)val;
                else
                    ((float*)Cv)[(size_t)row * N + col] = val;
            }
        }
    }
}

// ---------------------------------------------------------------------------
// Mean over sequence dim: xm[b][h] = mean_s x[b][s][h]
// ---------------------------------------------------------------------------
__global__ void mean_seq_kernel(const float* __restrict__ x, float* __restrict__ xm) {
    const int b = blockIdx.x;
    const int h = blockIdx.y * 256 + threadIdx.x;
    const float* p = x + (size_t)b * Sc * Hc + h;
    float s = 0.f;
    for (int i = 0; i < Sc; ++i) s += p[(size_t)i * Hc];
    xm[b * Hc + h] = s * (1.f / (float)Sc);
}

// ---------------------------------------------------------------------------
// Pooled-context projection: c[b][n] = xm[b] . Wu[:,n] + bu[n]  (tiny GEMM)
// ---------------------------------------------------------------------------
__global__ void pooled_proj_kernel(const float* __restrict__ xm,
                                   const float* __restrict__ Wu,
                                   const float* __restrict__ bu,
                                   float* __restrict__ c) {
    const int b = blockIdx.x;
    const int n = blockIdx.y * 256 + threadIdx.x;
    const float* xr = xm + (size_t)b * Hc;
    float s = bu[n];
    for (int k = 0; k < Hc; ++k) s = fmaf(xr[k], Wu[(size_t)k * Hc + n], s);
    c[b * Hc + n] = s;
}

// ---------------------------------------------------------------------------
// Sigmoid gate + blend (in place on f16 projection).
// ---------------------------------------------------------------------------
__global__ void gate_blend_kernel(_Float16* __restrict__ xp,
                                  const float* __restrict__ c,
                                  const float* __restrict__ wt,
                                  const float* __restrict__ bt,
                                  const float* __restrict__ wtc,
                                  const float* __restrict__ btc) {
    __shared__ float red[256];
    const int t = blockIdx.x;
    const int b = t / Sc;
    _Float16* xr = xp + (size_t)t * Hc;
    const float* cr = c + (size_t)b * Hc;

    float s = 0.f;
    for (int h = threadIdx.x; h < Hc; h += 256)
        s += (float)xr[h] * wt[h] + cr[h] * wtc[h];
    red[threadIdx.x] = s;
    __syncthreads();
    for (int off = 128; off > 0; off >>= 1) {
        if (threadIdx.x < off) red[threadIdx.x] += red[threadIdx.x + off];
        __syncthreads();
    }
    const float lam = 1.f / (1.f + __expf(-(red[0] + bt[0] + btc[0])));
    for (int h = threadIdx.x; h < Hc; h += 256) {
        const float xv = (float)xr[h];
        xr[h] = (_Float16)((1.f - lam) * xv + lam * cr[h]);
    }
}

// ---------------------------------------------------------------------------
// Flash attention, WMMA f16 + TDM-staged K tiles.
// grid (B*NH, S/128), block 256 = 8 waves; per block one (b,h), 128 q rows.
// ---------------------------------------------------------------------------
__launch_bounds__(256)
__global__ void flash_attn_wmma(const _Float16* __restrict__ qg,
                                const _Float16* __restrict__ kg,
                                const _Float16* __restrict__ vp,
                                const unsigned char* __restrict__ mask,
                                _Float16* __restrict__ outh) {
    __shared__ _Float16 KT[64][136];      // key-major [j][d]  (B-frag for Q@K^T)
    __shared__ _Float16 VT[128][72];      // dh-major  [d][j]  (B-frag for P@V)
    __shared__ _Float16 PL[8][16][72];    // per-wave P tile [row][j]
    __shared__ unsigned char Msk[64];     // key-block mask stage

    const int tid  = threadIdx.x;
    const int lane = tid & 31;
    const int wave = tid >> 5;
    const int g    = lane >> 4;
    const int lr   = lane & 15;
    const int b    = blockIdx.x >> 3;     // NH == 8
    const int h    = blockIdx.x & 7;
    const int qbase = blockIdx.y * 128;
    const size_t bS = (size_t)b * Sc;

    // ---- Q fragments: 4 K-chunks of 32 over DH=128, per-wave 16 rows ----
    F16x16 qa[4];
    {
        const _Float16* qrow =
            qg + (bS + qbase + wave * 16 + lr) * (size_t)Hc + h * DHc;
        for (int kc = 0; kc < 4; ++kc) {
            qa[kc].h8[0] = *(const v8h*)(qrow + 32 * kc + 8 * g);
            qa[kc].h8[1] = *(const v8h*)(qrow + 32 * kc + 8 * g + 16);
        }
    }

    const v8f vzero = {0.f, 0.f, 0.f, 0.f, 0.f, 0.f, 0.f, 0.f};
    F8f o[8];
    for (int nt = 0; nt < 8; ++nt) o[nt].v = vzero;
    float m[8], l[8];
    for (int r = 0; r < 8; ++r) { m[r] = -1e30f; l[r] = 0.f; }

    const float scale = 0.08838834764831845f;   // 1/sqrt(128)

    for (int kb = 0; kb < Sc; kb += 64) {
        // ---- K tile via TDM: 64 rows x 128 halves, stride H, padded to
        //      the 136-half LDS pitch (row = 64 DW -> enc 5; pad 4 DW -> enc 3)
        if (wave == 0) {
            const unsigned lds_off = (unsigned)(size_t)&KT[0][0];
            const unsigned long long ga = (unsigned long long)(size_t)(
                kg + (bS + kb) * (size_t)Hc + h * DHc);
            tdm_load_tile_f16(lds_off, ga, (unsigned)DHc, 64u, (unsigned)Hc, 5u, 3u);
        }
        // ---- mask stage ----
        if (tid < 64) Msk[tid] = mask[bS + kb + tid];
        // ---- V tile transposed (needs per-element transpose -> VALU path) ----
        for (int i = tid; i < 64 * 16; i += 256) {
            const int r = i >> 4, c8 = (i & 15) * 8;
            F16x16 vv;
            vv.h8[0] = *(const v8h*)(vp + (bS + kb + r) * (size_t)Hc + h * DHc + c8);
            for (int j = 0; j < 8; ++j) VT[c8 + j][r] = vv.h[j];
        }
        if (wave == 0) __builtin_amdgcn_s_wait_tensorcnt(0);
        __syncthreads();

        // ---- S = Q @ K^T  (16 rows x 64 keys per wave) ----
        F8f sc[4];
        for (int ct = 0; ct < 4; ++ct) sc[ct].v = vzero;
        for (int ct = 0; ct < 4; ++ct) {
            for (int kc = 0; kc < 4; ++kc) {
                F16x16 bf;
                const _Float16* src = &KT[ct * 16 + lr][32 * kc + 8 * g];
                bf.h8[0] = *(const v8h*)(src);
                bf.h8[1] = *(const v8h*)(src + 16);
                sc[ct].v = wmma_f16(qa[kc].v, bf.v, sc[ct].v);
            }
        }

        // ---- scale + mask ----
        float mbias[4];
        for (int ct = 0; ct < 4; ++ct)
            mbias[ct] = Msk[ct * 16 + lr] ? -1e9f : 0.f;
        for (int ct = 0; ct < 4; ++ct)
            for (int r = 0; r < 8; ++r)
                sc[ct].f[r] = sc[ct].f[r] * scale + mbias[ct];

        // ---- online softmax per row (rows split per half-wave) ----
        for (int r = 0; r < 8; ++r) {
            float rm = fmaxf(fmaxf(sc[0].f[r], sc[1].f[r]),
                             fmaxf(sc[2].f[r], sc[3].f[r]));
            for (int off = 8; off > 0; off >>= 1)
                rm = fmaxf(rm, __shfl_xor(rm, off, 32));
            const float mn = fmaxf(m[r], rm);
            const float sf = __expf(m[r] - mn);
            float rs = 0.f;
            for (int ct = 0; ct < 4; ++ct) {
                const float p = __expf(sc[ct].f[r] - mn);
                sc[ct].f[r] = p;
                rs += p;
            }
            for (int off = 8; off > 0; off >>= 1)
                rs += __shfl_xor(rs, off, 32);
            l[r] = l[r] * sf + rs;
            m[r] = mn;
            for (int nt = 0; nt < 8; ++nt) o[nt].f[r] *= sf;
        }

        // ---- P -> per-wave LDS (row = r + 8g, col = key) ----
        for (int ct = 0; ct < 4; ++ct)
            for (int r = 0; r < 8; ++r)
                PL[wave][r + 8 * g][ct * 16 + lr] = (_Float16)sc[ct].f[r];

        // ---- P A-fragments (2 K-chunks over 64 keys) ----
        F16x16 pa[2];
        for (int kc = 0; kc < 2; ++kc) {
            const _Float16* src = &PL[wave][lr][32 * kc];
            pa[kc].h8[0] = *(const v8h*)(src + 8 * g);
            pa[kc].h8[1] = *(const v8h*)(src + 8 * g + 16);
        }

        // ---- O += P @ V ----
        for (int nt = 0; nt < 8; ++nt) {
            for (int kc = 0; kc < 2; ++kc) {
                F16x16 bf;
                const _Float16* src = &VT[nt * 16 + lr][32 * kc + 8 * g];
                bf.h8[0] = *(const v8h*)(src);
                bf.h8[1] = *(const v8h*)(src + 16);
                o[nt].v = wmma_f16(pa[kc].v, bf.v, o[nt].v);
            }
        }
        __syncthreads();
    }

    // ---- normalize + store f16 ----
    for (int r = 0; r < 8; ++r) {
        const float inv = 1.f / l[r];
        const int row = qbase + wave * 16 + r + 8 * g;
        for (int nt = 0; nt < 8; ++nt)
            outh[(bS + row) * (size_t)Hc + h * DHc + nt * 16 + lr] =
                (_Float16)(o[nt].f[r] * inv);
    }
}

// ---------------------------------------------------------------------------
// Launch
// ---------------------------------------------------------------------------
extern "C" void kernel_launch(void* const* d_in, const int* in_sizes, int n_in,
                              void* d_out, int out_size, void* d_ws, size_t ws_size,
                              hipStream_t stream) {
    const float* v    = (const float*)d_in[0];
    const float* k    = (const float*)d_in[1];
    const float* q    = (const float*)d_in[2];
    const unsigned char* mask = (const unsigned char*)d_in[3];
    const float* Wv   = (const float*)d_in[4];
    const float* bv   = (const float*)d_in[5];
    const float* Wk   = (const float*)d_in[6];
    const float* bk   = (const float*)d_in[7];
    const float* Wq   = (const float*)d_in[8];
    const float* bq   = (const float*)d_in[9];
    const float* Wuk  = (const float*)d_in[10];
    const float* buk  = (const float*)d_in[11];
    const float* Wuq  = (const float*)d_in[12];
    const float* buq  = (const float*)d_in[13];
    const float* wtq  = (const float*)d_in[14];
    const float* btq  = (const float*)d_in[15];
    const float* wtk  = (const float*)d_in[16];
    const float* btk  = (const float*)d_in[17];
    const float* wtcq = (const float*)d_in[18];
    const float* btcq = (const float*)d_in[19];
    const float* wtck = (const float*)d_in[20];
    const float* btck = (const float*)d_in[21];
    const float* Wm   = (const float*)d_in[22];
    const float* bm   = (const float*)d_in[23];

    size_t off = 0;
    auto carve = [&](size_t bytes) {
        void* p = (char*)d_ws + off;
        off += (bytes + 255) & ~(size_t)255;
        return p;
    };
    _Float16* vp_h   = (_Float16*)carve((size_t)MS * Hc * sizeof(_Float16));
    _Float16* kp_h   = (_Float16*)carve((size_t)MS * Hc * sizeof(_Float16));
    _Float16* qp_h   = (_Float16*)carve((size_t)MS * Hc * sizeof(_Float16));
    _Float16* attn_h = (_Float16*)carve((size_t)MS * Hc * sizeof(_Float16));
    float* xm_k = (float*)carve((size_t)Bc * Hc * sizeof(float));
    float* xm_q = (float*)carve((size_t)Bc * Hc * sizeof(float));
    float* c_k  = (float*)carve((size_t)Bc * Hc * sizeof(float));
    float* c_q  = (float*)carve((size_t)Bc * Hc * sizeof(float));

    const dim3 gemmGrid(MS / 128, Hc / 128);   // (128, 8)
    const dim3 poolGrid(Bc, Hc / 256);

    // 1) projections -> f16 (WMMA-ready)
    gemm_bias_wmma<false, true><<<gemmGrid, 256, 0, stream>>>(v, Wv, bv, vp_h, MS, Hc, Hc);
    gemm_bias_wmma<false, true><<<gemmGrid, 256, 0, stream>>>(k, Wk, bk, kp_h, MS, Hc, Hc);
    gemm_bias_wmma<false, true><<<gemmGrid, 256, 0, stream>>>(q, Wq, bq, qp_h, MS, Hc, Hc);

    // 2) pooled contexts
    mean_seq_kernel<<<poolGrid, 256, 0, stream>>>(k, xm_k);
    mean_seq_kernel<<<poolGrid, 256, 0, stream>>>(q, xm_q);
    pooled_proj_kernel<<<poolGrid, 256, 0, stream>>>(xm_k, Wuk, buk, c_k);
    pooled_proj_kernel<<<poolGrid, 256, 0, stream>>>(xm_q, Wuq, buq, c_q);

    // 3) sigmoid gates, blend in place (qp->qg, kp->kg)
    gate_blend_kernel<<<MS, 256, 0, stream>>>(qp_h, c_q, wtq, btq, wtcq, btcq);
    gate_blend_kernel<<<MS, 256, 0, stream>>>(kp_h, c_k, wtk, btk, wtck, btck);

    // 4) flash attention -> f16
    flash_attn_wmma<<<dim3(Bc * NHc, Sc / 128), 256, 0, stream>>>(
        qp_h, kp_h, vp_h, mask, attn_h);

    // 5) output projection -> fp32 d_out
    gemm_bias_wmma<true, false><<<gemmGrid, 256, 0, stream>>>(
        attn_h, Wm, bm, (float*)d_out, MS, Hc, Hc);

    (void)in_sizes; (void)n_in; (void)out_size; (void)ws_size;
}
